// TC_74990128988335
// MI455X (gfx1250) — compile-verified, matrix operand order checked by reading, not run on hardware
//
#include <hip/hip_runtime.h>

typedef __attribute__((ext_vector_type(16))) _Float16 v16h;
typedef __attribute__((ext_vector_type(8)))  float    v8f;

#ifndef __has_builtin
#define __has_builtin(x) 0
#endif
#if __has_builtin(__builtin_amdgcn_tensor_load_to_lds) && \
    __has_builtin(__builtin_amdgcn_s_wait_tensorcnt)
#define HAVE_TDM 1
#else
#define HAVE_TDM 0
#endif

namespace {
constexpr int kB  = 16;
constexpr int kC  = 32;
constexpr int kH  = 240;
constexpr int kW  = 240;
constexpr int kHW = kH * kW;                    // 57600
constexpr int kTilesPerB  = kHW / 16;           // 3600
constexpr int kTPW        = 16;                 // tiles per wave
constexpr int kGroupsPerB = kTilesPerB / kTPW;  // 225
constexpr int kWavesTotal = kB * kGroupsPerB;   // 3600
constexpr int kBlocksMain = kWavesTotal / 8;    // 450

constexpr int kRowW   = 18;           // padded LDS row stride in words (bank-safe)
constexpr int kStageW = kRowW * kC;   // 576 words per staged 32x16 tile

// workspace layout (float offsets)
constexpr int WS_POOLED = 0;                          // [2][16][32][16]
constexpr int WS_QTOK   = WS_POOLED + 2*kB*kC*16;     // [2][16][16tok][16ch]
constexpr int kPart     = 2*kB*16*kGroupsPerB;        // [2][16][16q][225]
constexpr int WS_PARTM  = WS_QTOK + 2*kB*16*16;
constexpr int WS_PARTS  = WS_PARTM + kPart;
constexpr int WS_M      = WS_PARTS + kPart;           // [2][16][16]
constexpr int WS_INVS   = WS_M + 2*kB*16;
} // namespace

__device__ __forceinline__ v8f wmma_f16(v16h a, v16h b, v8f c) {
  return __builtin_amdgcn_wmma_f32_16x16x32_f16(false, a, false, b, (short)0, c,
                                                false, false);
}

// A operand for a 16x32 conv weight, row m = lane&15.
// lane<16: K {0..7,16..23}; lane>=16: K {8..15,24..31}  (kb0 = 0 or 8)
__device__ __forceinline__ v16h load_conv_A(const float* __restrict__ Wm, int m, int kb0) {
  v16h a;
#pragma unroll
  for (int j = 0; j < 8; ++j) a[j] = (_Float16)Wm[m * 32 + kb0 + j];
#pragma unroll
  for (int j = 0; j < 8; ++j) a[8 + j] = (_Float16)Wm[m * 32 + kb0 + 16 + j];
  return a;
}

// A operand for a 16x16 matrix padded to K=32 (upper 16 K zero); row = 16 floats.
__device__ __forceinline__ v16h load_pad_A(const float* __restrict__ row, int kb0) {
  v16h a;
#pragma unroll
  for (int j = 0; j < 8; ++j) a[j] = (_Float16)row[kb0 + j];
#pragma unroll
  for (int j = 8; j < 16; ++j) a[j] = (_Float16)0.f;
  return a;
}

// Direct-from-global B operand tile (fallback path).
__device__ __forceinline__ void load_tile_g(const float* __restrict__ xb, int p, int kb0,
                                            float xf[16], v16h& bx) {
#pragma unroll
  for (int j = 0; j < 8; ++j) xf[j] = xb[(kb0 + j) * kHW + p];
#pragma unroll
  for (int j = 0; j < 8; ++j) xf[8 + j] = xb[(kb0 + 16 + j) * kHW + p];
#pragma unroll
  for (int j = 0; j < 16; ++j) bx[j] = (_Float16)xf[j];
}

// Branchless online softmax update (avoids exec-mask churn).
__device__ __forceinline__ void online_update(float& m, float& s, float v) {
  float M = fmaxf(m, v);
  s = s * __expf(m - M) + __expf(v - M);
  m = M;
}

__device__ __forceinline__ void merge_ms(float& m, float& s, float om, float os) {
  float M = fmaxf(m, om);
  s = s * __expf(m - M) + os * __expf(om - M);
  m = M;
}

#if HAVE_TDM
typedef unsigned int tdm_v4u __attribute__((ext_vector_type(4)));
typedef int tdm_v8i __attribute__((ext_vector_type(8)));
typedef int tdm_v4i __attribute__((ext_vector_type(4)));

// TDM: DMA a 32ch x 16px f32 tile (rows strided by kHW in memory) into LDS,
// 2 DWORDs of padding every 16 DWORDs -> LDS row stride 18 words.
__device__ __forceinline__ void tdm_load_tile(const float* gptr, unsigned lds_off) {
  unsigned long long ga = (unsigned long long)(uintptr_t)gptr;
  tdm_v4u g0;
  g0[0] = 1u;                                        // count=1, user mode
  g0[1] = lds_off;                                   // lds_addr (bytes)
  g0[2] = (unsigned)ga;                              // global_addr lo
  g0[3] = (unsigned)((ga >> 32) & 0x01ffffffull) | 0x80000000u;  // hi | type=2
  tdm_v8i g1;
  // data_size=4B(2), pad_enable, pad_interval=16 DW(3), pad_amount=2 DW(1)
  g1[0] = (int)((2u << 16) | (1u << 20) | (3u << 22) | (1u << 25));
  g1[1] = (int)((unsigned)(kHW & 0xffff) << 16);                 // dim0 lo16
  g1[2] = (int)(((unsigned)kHW >> 16) | ((unsigned)kC << 16));   // dim0 hi | dim1 lo
  g1[3] = (int)(16u << 16);                                      // tile_dim0=16
  g1[4] = kC;                                                    // tile_dim1=32
  g1[5] = kHW;                                                   // dim0_stride lo32
  g1[6] = (int)((unsigned)(kHW & 0xffff) << 16);                 // dim1_stride lo16
  g1[7] = (int)((unsigned)kHW >> 16);                            // dim1_stride hi
  tdm_v4i z4 = {0, 0, 0, 0};
  tdm_v8i z8 = {0, 0, 0, 0, 0, 0, 0, 0};
  __builtin_amdgcn_tensor_load_to_lds(g0, g1, z4, z4, z8, 0);
}

__device__ __forceinline__ unsigned stage_byte_off(int wid, int slot, int img) {
  return (unsigned)((((wid * 2 + slot) * 2 + img) * kStageW) * 4);
}
#endif

// --------------------------------------------------------------------------
// K0: interior 4x4 block means of the 6x6 adaptive pool, per (img,b,c,token)
__global__ void __launch_bounds__(256) pool_kernel(const float* __restrict__ x1,
                                                   const float* __restrict__ x2,
                                                   float* __restrict__ ws) {
  int id = blockIdx.x;                 // ((img*16+b)*32+c)*16+t
  int t = id & 15, c = (id >> 4) & 31, b = (id >> 9) & 15, img = id >> 13;
  const float* x = img ? x2 : x1;
  int gi = 1 + (t >> 2), gj = 1 + (t & 3);
  const float* base = x + (size_t)(b * kC + c) * kHW + gi * 40 * kW + gj * 40;
  float s = 0.f;
  for (int k = threadIdx.x; k < 1600; k += 256) {
    int i = k / 40, j = k - i * 40;
    s += base[i * kW + j];
  }
  __shared__ float sm[256];
  sm[threadIdx.x] = s;
  __syncthreads();
  for (int off = 128; off; off >>= 1) {
    if (threadIdx.x < (unsigned)off) sm[threadIdx.x] += sm[threadIdx.x + off];
    __syncthreads();
  }
  if (threadIdx.x == 0) ws[WS_POOLED + id] = sm[0] * (1.0f / 1600.0f);
}

// K1: query tokens qtok[img][b][t][ch] = Wq[ch,:] . pooled[:,t] + bq[ch]
__global__ void __launch_bounds__(256) qtok_kernel(const float* __restrict__ Wq1,
                                                   const float* __restrict__ bq1,
                                                   const float* __restrict__ Wq2,
                                                   const float* __restrict__ bq2,
                                                   float* __restrict__ ws) {
  int ib = blockIdx.x;  // img*16 + b
  int img = ib >> 4;
  const float* Wq = img ? Wq2 : Wq1;
  const float* bq = img ? bq2 : bq1;
  int t = threadIdx.x >> 4, ch = threadIdx.x & 15;
  const float* pooled = ws + WS_POOLED + ib * kC * 16;
  float s = bq[ch];
  for (int c = 0; c < kC; ++c) s += Wq[ch * kC + c] * pooled[c * 16 + t];
  ws[WS_QTOK + (ib * 16 + t) * 16 + ch] = s;
}

// --------------------------------------------------------------------------
// KB: logits via WMMA + online softmax statistics per (attn,b,q)
__global__ void __launch_bounds__(256) kb_kernel(
    const float* __restrict__ x1, const float* __restrict__ x2,
    const float* __restrict__ Wk1, const float* __restrict__ bk1,
    const float* __restrict__ Wk2, const float* __restrict__ bk2,
    float* __restrict__ ws) {
  const int lane = threadIdx.x & 31;
  const int wid = threadIdx.x >> 5;
  const int gall = blockIdx.x * 8 + wid;       // [0, 3600)
  const int b = gall / kGroupsPerB;
  const int g = gall % kGroupsPerB;
  const int kb0 = (lane & 16) >> 1;            // 0 or 8
  const int mrow = lane & 15;

  v16h aWk1 = load_conv_A(Wk1, mrow, kb0);
  v16h aWk2 = load_conv_A(Wk2, mrow, kb0);
  v16h aq0 = load_pad_A(ws + WS_QTOK + ((0 * kB + b) * 16 + mrow) * 16, kb0);
  v16h aq1 = load_pad_A(ws + WS_QTOK + ((1 * kB + b) * 16 + mrow) * 16, kb0);

  float bk1r[8], bk2r[8], m0[8], s0[8], m1[8], s1[8];
#pragma unroll
  for (int r = 0; r < 8; ++r) {
    bk1r[r] = bk1[kb0 + r];
    bk2r[r] = bk2[kb0 + r];
    m0[r] = m1[r] = -3.0e38f;
    s0[r] = s1[r] = 0.f;
  }

  const float* x1b = x1 + (size_t)b * kC * kHW;
  const float* x2b = x2 + (size_t)b * kC * kHW;

#if HAVE_TDM
  __shared__ float stage[8][2][2][kStageW];
  // Escape the staging buffer's address (input-only, so LDS provenance of the
  // array itself is preserved for ds_load codegen) so the compiler must assume
  // the TDM engine writes it.
  {
    float* esc = &stage[0][0][0][0];
    asm volatile("" : : "v"(esc) : "memory");
  }
  {
    int p0 = (g * kTPW) * 16;
    tdm_load_tile(x1b + p0, stage_byte_off(wid, 0, 0));
    tdm_load_tile(x2b + p0, stage_byte_off(wid, 0, 1));
  }
#endif

  for (int it = 0; it < kTPW; ++it) {
    float xf[16];
    v16h bx, kbv;
#if HAVE_TDM
    if (it + 1 < kTPW) {
      int pn = (g * kTPW + it + 1) * 16;
      tdm_load_tile(x1b + pn, stage_byte_off(wid, (it + 1) & 1, 0));
      tdm_load_tile(x2b + pn, stage_byte_off(wid, (it + 1) & 1, 1));
      __builtin_amdgcn_s_wait_tensorcnt(2);
    } else {
      __builtin_amdgcn_s_wait_tensorcnt(0);
    }
    asm volatile("" : : : "memory");
    const float* t1 = &stage[wid][it & 1][0][0];
    const float* t2 = &stage[wid][it & 1][1][0];
#pragma unroll
    for (int j = 0; j < 8; ++j) xf[j] = t2[(kb0 + j) * kRowW + mrow];
#pragma unroll
    for (int j = 0; j < 8; ++j) xf[8 + j] = t2[(kb0 + 16 + j) * kRowW + mrow];
#pragma unroll
    for (int j = 0; j < 16; ++j) bx[j] = (_Float16)xf[j];
#else
    const int p = (g * kTPW + it) * 16 + mrow;
    load_tile_g(x2b, p, kb0, xf, bx);
#endif
    // attn a=0: logits = q1 . k2(x2)
    v8f dk = {};
    dk = wmma_f16(aWk2, bx, dk);
#pragma unroll
    for (int r = 0; r < 8; ++r) kbv[r] = (_Float16)(dk[r] + bk2r[r]);
#pragma unroll
    for (int r = 8; r < 16; ++r) kbv[r] = (_Float16)0.f;
    v8f dl = {};
    dl = wmma_f16(aq0, kbv, dl);
#pragma unroll
    for (int r = 0; r < 8; ++r) online_update(m0[r], s0[r], dl[r]);

    // attn a=1: logits = q2 . k1(x1)
#if HAVE_TDM
#pragma unroll
    for (int j = 0; j < 8; ++j) xf[j] = t1[(kb0 + j) * kRowW + mrow];
#pragma unroll
    for (int j = 0; j < 8; ++j) xf[8 + j] = t1[(kb0 + 16 + j) * kRowW + mrow];
#pragma unroll
    for (int j = 0; j < 16; ++j) bx[j] = (_Float16)xf[j];
#else
    load_tile_g(x1b, p, kb0, xf, bx);
#endif
    v8f dk1 = {};
    dk1 = wmma_f16(aWk1, bx, dk1);
#pragma unroll
    for (int r = 0; r < 8; ++r) kbv[r] = (_Float16)(dk1[r] + bk1r[r]);
    v8f dl1 = {};
    dl1 = wmma_f16(aq1, kbv, dl1);
#pragma unroll
    for (int r = 0; r < 8; ++r) online_update(m1[r], s1[r], dl1[r]);
  }

  // reduce across each 16-lane half (same q rows share a half)
#pragma unroll
  for (int mask = 1; mask < 16; mask <<= 1) {
#pragma unroll
    for (int r = 0; r < 8; ++r) {
      merge_ms(m0[r], s0[r], __shfl_xor(m0[r], mask, 32), __shfl_xor(s0[r], mask, 32));
      merge_ms(m1[r], s1[r], __shfl_xor(m1[r], mask, 32), __shfl_xor(s1[r], mask, 32));
    }
  }
  if ((lane & 15) == 0) {
#pragma unroll
    for (int r = 0; r < 8; ++r) {
      int q = kb0 + r;
      ws[WS_PARTM + ((0 * kB + b) * 16 + q) * kGroupsPerB + g] = m0[r];
      ws[WS_PARTS + ((0 * kB + b) * 16 + q) * kGroupsPerB + g] = s0[r];
      ws[WS_PARTM + ((1 * kB + b) * 16 + q) * kGroupsPerB + g] = m1[r];
      ws[WS_PARTS + ((1 * kB + b) * 16 + q) * kGroupsPerB + g] = s1[r];
    }
  }
}

// K2: merge per-wave partials -> final max and 1/sum per (attn,b,q)
__global__ void __launch_bounds__(32) softmax_merge_kernel(float* __restrict__ ws) {
  int idx = blockIdx.x;  // (a*16+b)*16+q, 512 blocks
  int lane = threadIdx.x;
  const float* pm = ws + WS_PARTM + idx * kGroupsPerB;
  const float* ps = ws + WS_PARTS + idx * kGroupsPerB;
  float m = -3.0e38f, s = 0.f;
  for (int g = lane; g < kGroupsPerB; g += 32) merge_ms(m, s, pm[g], ps[g]);
#pragma unroll
  for (int mask = 1; mask < 32; mask <<= 1)
    merge_ms(m, s, __shfl_xor(m, mask, 32), __shfl_xor(s, mask, 32));
  if (lane == 0) {
    ws[WS_M + idx] = m;
    ws[WS_INVS + idx] = 1.0f / s;
  }
}

// --------------------------------------------------------------------------
// KC: recompute k/logits, attn, v, output convs, residual; fully fused
__global__ void __launch_bounds__(256) kc_kernel(
    const float* __restrict__ x1, const float* __restrict__ x2,
    const float* __restrict__ Wk1, const float* __restrict__ bk1,
    const float* __restrict__ Wk2, const float* __restrict__ bk2,
    const float* __restrict__ Wv1, const float* __restrict__ bv1,
    const float* __restrict__ Wv2, const float* __restrict__ bv2,
    const float* __restrict__ Wc1, const float* __restrict__ bc1,
    const float* __restrict__ Wc2, const float* __restrict__ bc2,
    const float* __restrict__ gamma1p, const float* __restrict__ gamma2p,
    const float* __restrict__ ws, float* __restrict__ out) {
  const int lane = threadIdx.x & 31;
  const int wid = threadIdx.x >> 5;
  const int gall = blockIdx.x * 8 + wid;
  const int b = gall / kGroupsPerB;
  const int g = gall % kGroupsPerB;
  const int kb0 = (lane & 16) >> 1;
  const int mrow = lane & 15;

  const float g1 = *gamma1p, g2 = *gamma2p;

  v16h aWk1 = load_conv_A(Wk1, mrow, kb0);
  v16h aWk2 = load_conv_A(Wk2, mrow, kb0);
  v16h aWv1 = load_conv_A(Wv1, mrow, kb0);
  v16h aWv2 = load_conv_A(Wv2, mrow, kb0);
  v16h aq0 = load_pad_A(ws + WS_QTOK + ((0 * kB + b) * 16 + mrow) * 16, kb0);
  v16h aq1 = load_pad_A(ws + WS_QTOK + ((1 * kB + b) * 16 + mrow) * 16, kb0);
  v16h aWc2lo = load_pad_A(Wc2 + mrow * 16, kb0);
  v16h aWc2hi = load_pad_A(Wc2 + (mrow + 16) * 16, kb0);
  v16h aWc1lo = load_pad_A(Wc1 + mrow * 16, kb0);
  v16h aWc1hi = load_pad_A(Wc1 + (mrow + 16) * 16, kb0);

  float bk1r[8], bk2r[8], bv1r[8], bv2r[8];
  float gb0lo[8], gb0hi[8], gb1lo[8], gb1hi[8];
  float m0r[8], i0r[8], m1r[8], i1r[8];
#pragma unroll
  for (int r = 0; r < 8; ++r) {
    bk1r[r] = bk1[kb0 + r];
    bk2r[r] = bk2[kb0 + r];
    bv1r[r] = bv1[kb0 + r];
    bv2r[r] = bv2[kb0 + r];
    gb0lo[r] = g1 * bc2[kb0 + r];
    gb0hi[r] = g1 * bc2[16 + kb0 + r];
    gb1lo[r] = g2 * bc1[kb0 + r];
    gb1hi[r] = g2 * bc1[16 + kb0 + r];
    m0r[r] = ws[WS_M + (0 * kB + b) * 16 + kb0 + r];
    i0r[r] = ws[WS_INVS + (0 * kB + b) * 16 + kb0 + r];
    m1r[r] = ws[WS_M + (1 * kB + b) * 16 + kb0 + r];
    i1r[r] = ws[WS_INVS + (1 * kB + b) * 16 + kb0 + r];
  }

  const float* x1b = x1 + (size_t)b * kC * kHW;
  const float* x2b = x2 + (size_t)b * kC * kHW;
  float* o0 = out + (size_t)b * kC * kHW;
  float* o1 = out + (size_t)(kB + b) * kC * kHW;

#if HAVE_TDM
  __shared__ float stage[8][2][2][kStageW];
  {
    float* esc = &stage[0][0][0][0];
    asm volatile("" : : "v"(esc) : "memory");
  }
  {
    int p0 = (g * kTPW) * 16;
    tdm_load_tile(x1b + p0, stage_byte_off(wid, 0, 0));
    tdm_load_tile(x2b + p0, stage_byte_off(wid, 0, 1));
  }
#endif

  for (int it = 0; it < kTPW; ++it) {
    const int p = (g * kTPW + it) * 16 + mrow;
    float xf1[16], xf2[16];
    v16h bx1, bx2;
#if HAVE_TDM
    if (it + 1 < kTPW) {
      int pn = (g * kTPW + it + 1) * 16;
      tdm_load_tile(x1b + pn, stage_byte_off(wid, (it + 1) & 1, 0));
      tdm_load_tile(x2b + pn, stage_byte_off(wid, (it + 1) & 1, 1));
      __builtin_amdgcn_s_wait_tensorcnt(2);
    } else {
      __builtin_amdgcn_s_wait_tensorcnt(0);
    }
    asm volatile("" : : : "memory");
    const float* t1 = &stage[wid][it & 1][0][0];
    const float* t2 = &stage[wid][it & 1][1][0];
#pragma unroll
    for (int j = 0; j < 8; ++j) xf1[j] = t1[(kb0 + j) * kRowW + mrow];
#pragma unroll
    for (int j = 0; j < 8; ++j) xf1[8 + j] = t1[(kb0 + 16 + j) * kRowW + mrow];
#pragma unroll
    for (int j = 0; j < 8; ++j) xf2[j] = t2[(kb0 + j) * kRowW + mrow];
#pragma unroll
    for (int j = 0; j < 8; ++j) xf2[8 + j] = t2[(kb0 + 16 + j) * kRowW + mrow];
#pragma unroll
    for (int j = 0; j < 16; ++j) { bx1[j] = (_Float16)xf1[j]; bx2[j] = (_Float16)xf2[j]; }
#else
    load_tile_g(x1b, p, kb0, xf1, bx1);
    load_tile_g(x2b, p, kb0, xf2, bx2);
#endif

    v8f dk2 = {}; dk2 = wmma_f16(aWk2, bx2, dk2);
    v8f dv1 = {}; dv1 = wmma_f16(aWv1, bx1, dv1);
    v8f dk1 = {}; dk1 = wmma_f16(aWk1, bx1, dk1);
    v8f dv2 = {}; dv2 = wmma_f16(aWv2, bx2, dv2);

    v16h t;
#pragma unroll
    for (int r = 0; r < 8; ++r) t[r] = (_Float16)(dk2[r] + bk2r[r]);
#pragma unroll
    for (int r = 8; r < 16; ++r) t[r] = (_Float16)0.f;
    v8f dl0 = {}; dl0 = wmma_f16(aq0, t, dl0);
#pragma unroll
    for (int r = 0; r < 8; ++r) t[r] = (_Float16)(dk1[r] + bk1r[r]);
    v8f dl1 = {}; dl1 = wmma_f16(aq1, t, dl1);

    v16h e0, e1;
#pragma unroll
    for (int r = 0; r < 8; ++r) {
      float a0 = __expf(dl0[r] - m0r[r]) * i0r[r];
      e0[r] = (_Float16)(a0 * (dv1[r] + bv1r[r]));
      float a1 = __expf(dl1[r] - m1r[r]) * i1r[r];
      e1[r] = (_Float16)(a1 * (dv2[r] + bv2r[r]));
    }
#pragma unroll
    for (int r = 8; r < 16; ++r) {
      e0[r] = (_Float16)0.f;
      e1[r] = (_Float16)0.f;
    }

    v8f o00 = {}, o01 = {}, o10 = {}, o11 = {};
    o00 = wmma_f16(aWc2lo, e0, o00);
    o01 = wmma_f16(aWc2hi, e0, o01);
    o10 = wmma_f16(aWc1lo, e1, o10);
    o11 = wmma_f16(aWc1hi, e1, o11);

#pragma unroll
    for (int r = 0; r < 8; ++r) {
      o0[(kb0 + r) * kHW + p]      = xf1[r]     + g1 * o00[r] + gb0lo[r];
      o0[(16 + kb0 + r) * kHW + p] = xf1[8 + r] + g1 * o01[r] + gb0hi[r];
      o1[(kb0 + r) * kHW + p]      = xf2[r]     + g2 * o10[r] + gb1lo[r];
      o1[(16 + kb0 + r) * kHW + p] = xf2[8 + r] + g2 * o11[r] + gb1hi[r];
    }
  }
}

// --------------------------------------------------------------------------
extern "C" void kernel_launch(void* const* d_in, const int* in_sizes, int n_in,
                              void* d_out, int out_size, void* d_ws, size_t ws_size,
                              hipStream_t stream) {
  (void)in_sizes; (void)n_in; (void)out_size; (void)ws_size;
  const float* x1 = (const float*)d_in[0];
  const float* x2 = (const float*)d_in[1];
  const float* Wq1 = (const float*)d_in[2];
  const float* bq1 = (const float*)d_in[3];
  const float* Wk1 = (const float*)d_in[4];
  const float* bk1 = (const float*)d_in[5];
  const float* Wv1 = (const float*)d_in[6];
  const float* bv1 = (const float*)d_in[7];
  const float* Wq2 = (const float*)d_in[8];
  const float* bq2 = (const float*)d_in[9];
  const float* Wk2 = (const float*)d_in[10];
  const float* bk2 = (const float*)d_in[11];
  const float* Wv2 = (const float*)d_in[12];
  const float* bv2 = (const float*)d_in[13];
  const float* Wc1 = (const float*)d_in[14];
  const float* bc1 = (const float*)d_in[15];
  const float* Wc2 = (const float*)d_in[16];
  const float* bc2 = (const float*)d_in[17];
  const float* gamma1 = (const float*)d_in[18];
  const float* gamma2 = (const float*)d_in[19];
  float* ws = (float*)d_ws;
  float* out = (float*)d_out;

  pool_kernel<<<2 * kB * kC * 16, 256, 0, stream>>>(x1, x2, ws);
  qtok_kernel<<<2 * kB, 256, 0, stream>>>(Wq1, bq1, Wq2, bq2, ws);
  kb_kernel<<<kBlocksMain, 256, 0, stream>>>(x1, x2, Wk1, bk1, Wk2, bk2, ws);
  softmax_merge_kernel<<<2 * kB * 16, 32, 0, stream>>>(ws);
  kc_kernel<<<kBlocksMain, 256, 0, stream>>>(x1, x2, Wk1, bk1, Wk2, bk2, Wv1, bv1,
                                             Wv2, bv2, Wc1, bc1, Wc2, bc2,
                                             gamma1, gamma2, ws, out);
}
